// LlamaGroupedQueryAttention_14319420965580
// MI455X (gfx1250) — matrix operand order repaired and checked by reading, not compiled, not run on hardware
//
#include <hip/hip_runtime.h>
#include <cstdint>
#include <cstddef>

typedef __attribute__((ext_vector_type(16))) __bf16 v16bf;
typedef __attribute__((ext_vector_type(8)))  float  v8f;

typedef unsigned int tdm_v4u __attribute__((ext_vector_type(4)));
typedef int          tdm_v8i __attribute__((ext_vector_type(8)));
typedef int          tdm_v4i __attribute__((ext_vector_type(4)));

#define S_LEN 2048
#define NHEADS 32
#define HD 64
#define SCALE 0.125f
#define NEG_INF (-1.0e9f)

__device__ __forceinline__ unsigned short f32_to_bf16(float f) {
  union { float f; unsigned int u; } x; x.f = f;
  unsigned int r = x.u + 0x7fffu + ((x.u >> 16) & 1u);
  return (unsigned short)(r >> 16);
}

union ABFrag { v16bf v; unsigned int u[8]; };

// A-matrix 16x32 bf16 fragment (ISA 7.12.2): lane holds row (lane&15);
// VGPR p, slot t -> K = 2*(p&3) + t + 8*half + 16*(p>>2)
__device__ __forceinline__ v16bf load_fragA(const unsigned short* base, int stride,
                                            int row, int k0, int half) {
  ABFrag f;
#pragma unroll
  for (int p = 0; p < 8; ++p) {
    int k = k0 + ((p & 3) << 1) + (half << 3) + ((p >> 2) << 4);
    f.u[p] = *(const unsigned int*)(base + (size_t)row * stride + k);
  }
  return f.v;
}

// B-matrix 32x16 bf16 fragment: lane holds col (lane&15);
// VGPR p, slot t -> K = 2*(p&3) + t + 8*half + 16*(p>>2)
__device__ __forceinline__ v16bf load_fragB(const unsigned short* base, int stride,
                                            int col, int k0, int half) {
  ABFrag f;
#pragma unroll
  for (int p = 0; p < 8; ++p) {
    int k = k0 + ((p & 3) << 1) + (half << 3) + ((p >> 2) << 4);
    f.u[p] = *(const unsigned int*)(base + (size_t)col * stride + k);
  }
  return f.v;
}

__global__ void k_f32_to_bf16(const float* __restrict__ src,
                              unsigned short* __restrict__ dst, int n) {
  int i = blockIdx.x * blockDim.x + threadIdx.x;
  if (i < n) dst[i] = f32_to_bf16(src[i]);
}

// C(MxN) = A(MxK) @ B(NxK)^T ; A,B bf16 row-major, C fp32.
// block = 128 threads (4 waves); block tile = 16(M) x 64(N).
// A stripe (16 x K bf16) staged into LDS via async global->LDS b128 copies.
__global__ void k_gemm_bf16(const unsigned short* __restrict__ A,
                            const unsigned short* __restrict__ B,
                            float* __restrict__ C, int M, int N, int K) {
  extern __shared__ unsigned char smem[];
  unsigned short* As = (unsigned short*)smem;          // 16 x K bf16 @ LDS offset 0
  const int m0 = blockIdx.x * 16;
  const int nb = blockIdx.y * 64;

  // stage A stripe (rows m0..m0+15 contiguous): 16B per async op per lane
  {
    const int chunks = (16 * K) / 8;                   // uint4 = 8 bf16
    const uint4* Ag = (const uint4*)(A + (size_t)m0 * K);
    for (int c = threadIdx.x; c < chunks; c += blockDim.x) {
      unsigned lds_off = (unsigned)(c * 16);           // dyn-LDS base == offset 0
      unsigned long long ga = (unsigned long long)(uintptr_t)(Ag + c);
      asm volatile("global_load_async_to_lds_b128 %0, %1, off"
                   :: "v"(lds_off), "v"(ga) : "memory");
    }
    asm volatile("s_wait_asynccnt 0" ::: "memory");
  }
  __syncthreads();

  const int wave = threadIdx.x >> 5;
  const int lane = threadIdx.x & 31;
  const int half = lane >> 4;
  const int n0 = nb + wave * 16;
  const int nl = lane & 15;

  v8f acc = {};
  for (int k0 = 0; k0 < K; k0 += 32) {
    __builtin_prefetch(B + (size_t)(n0 + nl) * K + k0 + 128, 0, 3);
    v16bf a = load_fragA(As, K, nl, k0, half);
    v16bf b = load_fragB(B, K, n0 + nl, k0, half);
    acc = __builtin_amdgcn_wmma_f32_16x16x32_bf16(false, a, false, b,
                                                  (short)0, acc, false, false);
  }
  const int col = n0 + nl;
#pragma unroll
  for (int i = 0; i < 8; ++i) {
    int row = m0 + i + (half << 3);
    C[(size_t)row * N + col] = acc[i];
  }
}

// RoPE on q,k + pack everything to bf16. v is written transposed (512 x S)
// so the P@V B-fragment gets contiguous-K loads.
__global__ void k_rope_pack(const float* __restrict__ qkv,
                            const float* __restrict__ cosb,
                            const float* __restrict__ sinb,
                            unsigned short* __restrict__ qb,
                            unsigned short* __restrict__ kb,
                            unsigned short* __restrict__ vtb) {
  int idx = blockIdx.x * blockDim.x + threadIdx.x;
  if (idx >= S_LEN * 3072) return;
  int s = idx / 3072;
  int c = idx - s * 3072;
  float x = qkv[idx];
  if (c < 2048) {                       // q
    int d = c & 63;
    float rot = (d < 32) ? -qkv[idx + 32] : qkv[idx - 32];
    float v = x * cosb[s * 64 + d] + rot * sinb[s * 64 + d];
    qb[(size_t)s * 2048 + c] = f32_to_bf16(v);
  } else if (c < 2560) {                // k
    int cc = c - 2048;
    int d = cc & 63;
    float rot = (d < 32) ? -qkv[idx + 32] : qkv[idx - 32];
    float v = x * cosb[s * 64 + d] + rot * sinb[s * 64 + d];
    kb[(size_t)s * 512 + cc] = f32_to_bf16(v);
  } else {                              // v -> transposed
    int cc = c - 2560;
    vtb[(size_t)cc * 2048 + s] = f32_to_bf16(x);
  }
}

// LDS layout for k_attn (dynamic shared, base offset 0):
#define ATT_SC_OFF   0                       // 16 x 2048 f32  (131072 B)
#define ATT_PB_OFF   (16 * 2048 * 4)         // 16 x 2048 bf16 ( 65536 B)
#define ATT_RED_OFF  (ATT_PB_OFF + 16 * 2048 * 2)   // 16 x 8 f32 (512 B)
#define ATT_QT_OFF   (ATT_RED_OFF + 512)     // 16 x 64 bf16   (  2048 B)
#define ATT_LDS_SIZE (ATT_QT_OFF + 2048)

// One workgroup = (head, 16 q-rows). q tile staged to LDS by the Tensor Data
// Mover; scores -> LDS fp32 -> softmax -> fp32 attn_weights + bf16 probs in
// LDS -> P@V -> bf16 attn for out-proj.
__global__ void k_attn(const unsigned short* __restrict__ qb,
                       const unsigned short* __restrict__ kb,
                       const unsigned short* __restrict__ vtb,
                       float* __restrict__ attnw,
                       unsigned short* __restrict__ attnb) {
  extern __shared__ unsigned char smem[];
  float* sc = (float*)(smem + ATT_SC_OFF);
  unsigned short* pb = (unsigned short*)(smem + ATT_PB_OFF);
  float* red = (float*)(smem + ATT_RED_OFF);
  unsigned short* qt = (unsigned short*)(smem + ATT_QT_OFF);

  const int q0 = blockIdx.x * 16;
  const int h = blockIdx.y;
  const int hkv = h >> 2;               // N_REP = 4
  const int wave = threadIdx.x >> 5;
  const int lane = threadIdx.x & 31;
  const int half = lane >> 4;
  const int nl = lane & 15;

#if __has_builtin(__builtin_amdgcn_tensor_load_to_lds)
  // ---- TDM: stage 16x64 bf16 q tile (tile 16 rows x 128B, pitch 4096B) ----
  if (threadIdx.x == 0) {               // TDM ignores EXEC; one wave issues once
    unsigned long long ga =
        (unsigned long long)(uintptr_t)(qb + (size_t)q0 * 2048 + h * HD);
    tdm_v4u g0;
    g0.x = 1u;                                   // count = 1 valid descriptor
    g0.y = (unsigned)ATT_QT_OFF;                 // lds_addr
    g0.z = (unsigned)(ga & 0xffffffffu);         // global_addr[31:0]
    g0.w = (unsigned)((ga >> 32) & 0x01ffffffu) | (2u << 30);  // addr[56:32]|type=2
    tdm_v8i g1;
    g1[0] = (int)(3u << 16);                     // wg_mask=0, data_size=8B
    g1[1] = (int)(16u << 16);                    // tensor_dim0 = 16 (8B units)
    g1[2] = (int)(16u << 16);                    // tensor_dim1 = 16
    g1[3] = (int)(16u << 16);                    // tile_dim0 = 16 (=128B row)
    g1[4] = 16;                                  // tile_dim1 = 16 rows
    g1[5] = 512;                                 // tensor_dim0_stride = 4096B
    g1[6] = 0;
    g1[7] = 0;
    tdm_v4i z4 = {0, 0, 0, 0};
    tdm_v8i z8 = {0, 0, 0, 0, 0, 0, 0, 0};
    __builtin_amdgcn_tensor_load_to_lds(g0, g1, z4, z4, z8, 0);
    __builtin_amdgcn_s_wait_tensorcnt(0);
  }
  __syncthreads();
  v16bf qa0 = load_fragA(qt, 64, nl, 0, half);
  v16bf qa1 = load_fragA(qt, 64, nl, 32, half);
#else
  const unsigned short* qbase = qb + (size_t)q0 * 2048 + h * HD;
  v16bf qa0 = load_fragA(qbase, 2048, nl, 0, half);
  v16bf qa1 = load_fragA(qbase, 2048, nl, 32, half);
#endif

  // ---- scores = q @ k^T (per-head K-dim = HD = 64 -> 2 WMMA per tile) ----
  for (int kt = wave; kt < 128; kt += 4) {
    const unsigned short* kbase = kb + (size_t)kt * 16 * 512 + hkv * HD;
    v16bf b0 = load_fragB(kbase, 512, nl, 0, half);
    v16bf b1 = load_fragB(kbase, 512, nl, 32, half);
    v8f acc = {};
    acc = __builtin_amdgcn_wmma_f32_16x16x32_bf16(false, qa0, false, b0,
                                                  (short)0, acc, false, false);
    acc = __builtin_amdgcn_wmma_f32_16x16x32_bf16(false, qa1, false, b1,
                                                  (short)0, acc, false, false);
    const int skl = kt * 16 + nl;
#pragma unroll
    for (int i = 0; i < 8; ++i) {
      int r = i + (half << 3);
      float v = acc[i] * SCALE;
      if (skl > q0 + r) v += NEG_INF;   // causal mask
      sc[r * 2048 + skl] = v;
    }
  }
  __syncthreads();

  // ---- row softmax: 16 rows x 2048, 8 threads per row ----
  const int row = threadIdx.x >> 3;
  const int sub = threadIdx.x & 7;
  float* rp = sc + row * 2048 + sub * 256;
  float mx = -3.0e38f;
  for (int j = 0; j < 256; ++j) mx = fmaxf(mx, rp[j]);
  red[row * 8 + sub] = mx;
  __syncthreads();
  mx = red[row * 8];
  for (int j = 1; j < 8; ++j) mx = fmaxf(mx, red[row * 8 + j]);
  __syncthreads();
  float sum = 0.f;
  for (int j = 0; j < 256; ++j) { float e = __expf(rp[j] - mx); rp[j] = e; sum += e; }
  red[row * 8 + sub] = sum;
  __syncthreads();
  sum = 0.f;
  for (int j = 0; j < 8; ++j) sum += red[row * 8 + j];
  const float inv = 1.0f / sum;

  float* wdst = attnw + ((size_t)h * 2048 + q0 + row) * 2048 + sub * 256;
  unsigned short* pdst = pb + row * 2048 + sub * 256;
  for (int j = 0; j < 256; j += 4) {
    float4 pv;
    pv.x = rp[j] * inv; pv.y = rp[j + 1] * inv;
    pv.z = rp[j + 2] * inv; pv.w = rp[j + 3] * inv;
    *(float4*)(wdst + j) = pv;                       // fp32 attn_weights output
    pdst[j] = f32_to_bf16(pv.x); pdst[j + 1] = f32_to_bf16(pv.y);
    pdst[j + 2] = f32_to_bf16(pv.z); pdst[j + 3] = f32_to_bf16(pv.w);
  }
  __syncthreads();

  // ---- attn = P @ V : wave owns 16 head-dims, K = 2048 -> 64 WMMA ----
  const int n0 = wave * 16;
  const unsigned short* vbase = vtb + (size_t)(hkv * HD) * 2048;
  v8f acc = {};
  for (int k0 = 0; k0 < 2048; k0 += 32) {
    v16bf a = load_fragA(pb, 2048, nl, k0, half);
    v16bf b = load_fragB(vbase, 2048, n0 + nl, k0, half);
    acc = __builtin_amdgcn_wmma_f32_16x16x32_bf16(false, a, false, b,
                                                  (short)0, acc, false, false);
  }
  const int col = h * HD + n0 + nl;
#pragma unroll
  for (int i = 0; i < 8; ++i) {
    int r = q0 + i + (half << 3);
    attnb[(size_t)r * 2048 + col] = f32_to_bf16(acc[i]);
  }
}

extern "C" void kernel_launch(void* const* d_in, const int* in_sizes, int n_in,
                              void* d_out, int out_size, void* d_ws, size_t ws_size,
                              hipStream_t stream) {
  (void)in_sizes; (void)n_in; (void)out_size; (void)ws_size;
  const float* hs   = (const float*)d_in[0];
  const float* cosb = (const float*)d_in[1];
  const float* sinb = (const float*)d_in[2];
  // d_in[3] = attention_mask (causal; applied analytically in-kernel)
  const float* Wq = (const float*)d_in[4];
  const float* Wk = (const float*)d_in[5];
  const float* Wv = (const float*)d_in[6];
  const float* Wo = (const float*)d_in[7];

  float* outp  = (float*)d_out;                       // (2048 x 2048)
  float* attnw = outp + (size_t)2048 * 2048;          // (32 x 2048 x 2048)

  unsigned char* w = (unsigned char*)d_ws;
  unsigned short* hsb   = (unsigned short*)w; w += (size_t)2048 * 2048 * 2;
  unsigned short* wb    = (unsigned short*)w; w += (size_t)3072 * 2048 * 2;
  unsigned short* wob   = (unsigned short*)w; w += (size_t)2048 * 2048 * 2;
  float*          qkv   = (float*)w;          w += (size_t)2048 * 3072 * 4;
  unsigned short* qb    = (unsigned short*)w; w += (size_t)2048 * 2048 * 2;
  unsigned short* kb    = (unsigned short*)w; w += (size_t)2048 * 512  * 2;
  unsigned short* vtb   = (unsigned short*)w; w += (size_t)512  * 2048 * 2;
  unsigned short* attnb = (unsigned short*)w; w += (size_t)2048 * 2048 * 2;

  int n;
  n = 2048 * 2048; k_f32_to_bf16<<<(n + 255) / 256, 256, 0, stream>>>(hs, hsb, n);
  n = 2048 * 2048; k_f32_to_bf16<<<(n + 255) / 256, 256, 0, stream>>>(Wq, wb, n);
  n = 512 * 2048;  k_f32_to_bf16<<<(n + 255) / 256, 256, 0, stream>>>(Wk, wb + (size_t)2048 * 2048, n);
  n = 512 * 2048;  k_f32_to_bf16<<<(n + 255) / 256, 256, 0, stream>>>(Wv, wb + (size_t)2560 * 2048, n);
  n = 2048 * 2048; k_f32_to_bf16<<<(n + 255) / 256, 256, 0, stream>>>(Wo, wob, n);

  // fused QKV projection: C(2048x3072) = hs @ [Wq;Wk;Wv]^T
  k_gemm_bf16<<<dim3(128, 48), 128, 16 * 2048 * 2, stream>>>(hsb, wb, qkv, 2048, 3072, 2048);

  n = 2048 * 3072; k_rope_pack<<<(n + 255) / 256, 256, 0, stream>>>(qkv, cosb, sinb, qb, kb, vtb);

  // attention: 128 q-tiles x 32 heads
  k_attn<<<dim3(128, 32), 128, ATT_LDS_SIZE, stream>>>(qb, kb, vtb, attnw, attnb);

  // output projection: out = attn @ Wo^T
  k_gemm_bf16<<<dim3(128, 32), 128, 16 * 2048 * 2, stream>>>(attnb, wob, outp, 2048, 2048, 2048);
}
